// SelfAttention_37383395345094
// MI455X (gfx1250) — matrix operand order repaired and checked by reading, not compiled, not run on hardware
//
#include <hip/hip_runtime.h>
#include <hip/hip_bf16.h>

// ---------------------------------------------------------------------------
// Self-attention for MI455X (gfx1250, wave32, WMMA bf16 16x16x32, f32 accum)
// B=2, T=2048, C=1024, H=16, d=64
// Round 2: bigger register tiles (GEMM 32x64/wave, attention 32 Q-rows/wave)
// to raise WMMA:VMEM ratio per roofline; rel_pos prefetch.
// ---------------------------------------------------------------------------

#define B_  2
#define T_  2048
#define C_  1024
#define H_  16
#define D_  64
#define M_  (B_ * T_)          // 4096 rows for the projections

typedef __bf16 bf16_t;
typedef bf16_t v16bf __attribute__((ext_vector_type(16)));
typedef bf16_t v8bf  __attribute__((ext_vector_type(8)));
typedef float  v8f   __attribute__((ext_vector_type(8)));

#define NEG_BIG (-1e30f)

// ---------------------------------------------------------------------------
// Fragment loaders (this code's contiguous k order, same for A and B; wave32)
// A 16x32 bf16: lane l -> row m = m0+(l&15); koff = (l>>4)*16;
//   elems 0..7  -> K = k0+koff+0..7        (16B contiguous)
//   elems 8..15 -> K = k0+koff+8..15       (16B contiguous)
// ---------------------------------------------------------------------------
__device__ __forceinline__ v16bf load_a_frag(const bf16_t* __restrict__ base,
                                             int ld, int m0, int k0, int lane) {
  const int m    = m0 + (lane & 15);
  const int koff = (lane >> 4) * 16;
  const bf16_t* p = base + (size_t)m * ld + k0;
  union { v16bf v; v8bf h[2]; } u;
  u.h[0] = *(const v8bf*)(p + koff);
  u.h[1] = *(const v8bf*)(p + 8 + koff);
  return u.v;
}

// B 32x16 bf16, where "src" is row-major [N][K] (row n supplies B column n):
// lane l -> col n = n0+(l&15); elems i -> K = k0+(l>>4)*16 + i  (32B contiguous)
__device__ __forceinline__ v16bf load_b_frag(const bf16_t* __restrict__ base,
                                             int ld, int n0, int k0, int lane) {
  const int n  = n0 + (lane & 15);
  const int kb = k0 + (lane >> 4) * 16;
  return *(const v16bf*)(base + (size_t)n * ld + kb);
}

__device__ __forceinline__ v8f wmma_bf16(v16bf a, v16bf b, v8f c) {
  return __builtin_amdgcn_wmma_f32_16x16x32_bf16(false, a, false, b,
                                                 (short)0, c, false, false);
}

__device__ __forceinline__ v8f v8f_zero() {
  v8f z = {0.f, 0.f, 0.f, 0.f, 0.f, 0.f, 0.f, 0.f};
  return z;
}

// ---------------------------------------------------------------------------
// fp32 -> bf16 conversion (grid-strided)
// ---------------------------------------------------------------------------
__global__ void cvt_f32_bf16(const float* __restrict__ in,
                             bf16_t* __restrict__ out, int n) {
  for (int i = blockIdx.x * blockDim.x + threadIdx.x; i < n;
       i += gridDim.x * blockDim.x)
    out[i] = (bf16_t)in[i];
}

// ---------------------------------------------------------------------------
// QKV projection GEMM:  y = x @ W^T + b   (M=4096, N=1024, K=1024)
// blockDim = 128 (4 waves); each wave computes a 32(M) x 64(N) tile:
// 2 A-frags + 4 B-frags -> 8 WMMAs per K-step of 32  (~21 FLOP/byte).
// grid: (N/64, M/128, 3)  z selects Q/K/V.
// Q,K written as [b,h,t,d] bf16; V written transposed [b,h,d,t] bf16.
// ---------------------------------------------------------------------------
__global__ __launch_bounds__(128)
void qkv_gemm_kernel(const bf16_t* __restrict__ xb,
                     const bf16_t* __restrict__ wq, const bf16_t* __restrict__ wk,
                     const bf16_t* __restrict__ wv,
                     const float* __restrict__ bq, const float* __restrict__ bk,
                     const float* __restrict__ bv,
                     bf16_t* __restrict__ qb, bf16_t* __restrict__ kb,
                     bf16_t* __restrict__ vtb) {
  const int lane = threadIdx.x & 31;
  const int wave = threadIdx.x >> 5;
  const int proj = blockIdx.z;
  const bf16_t* W    = (proj == 0) ? wq : (proj == 1) ? wk : wv;
  const float*  bias = (proj == 0) ? bq : (proj == 1) ? bk : bv;

  const int n0 = blockIdx.x * 64;                 // 64 output columns
  const int m0 = blockIdx.y * 128 + wave * 32;    // 32 output rows

  v8f acc[2][4];
#pragma unroll
  for (int mi = 0; mi < 2; ++mi)
#pragma unroll
    for (int nj = 0; nj < 4; ++nj) acc[mi][nj] = v8f_zero();

  for (int k0 = 0; k0 < C_; k0 += 32) {
    v16bf a0 = load_a_frag(xb, C_, m0,      k0, lane);
    v16bf a1 = load_a_frag(xb, C_, m0 + 16, k0, lane);
#pragma unroll
    for (int nj = 0; nj < 4; ++nj) {
      v16bf b = load_b_frag(W, C_, n0 + nj * 16, k0, lane);
      acc[0][nj] = wmma_bf16(a0, b, acc[0][nj]);
      acc[1][nj] = wmma_bf16(a1, b, acc[1][nj]);
    }
  }

#pragma unroll
  for (int nj = 0; nj < 4; ++nj) {
    const int n  = n0 + nj * 16 + (lane & 15);
    const float bval = bias[n];
    const int h  = n >> 6;          // n / 64
    const int dd = n & 63;
#pragma unroll
    for (int mi = 0; mi < 2; ++mi) {
#pragma unroll
      for (int r = 0; r < 8; ++r) {
        const int m  = m0 + mi * 16 + r + 8 * (lane >> 4);
        const int bb = m >> 11;     // m / T
        const int t  = m & (T_ - 1);
        const float val = acc[mi][nj][r] + bval;
        if (proj == 2) {
          vtb[(((size_t)bb * H_ + h) * D_ + dd) * T_ + t] = (bf16_t)val;
        } else {
          bf16_t* dst = (proj == 0) ? qb : kb;
          dst[(((size_t)bb * H_ + h) * T_ + t) * D_ + dd] = (bf16_t)val;
        }
      }
    }
  }
}

// ---------------------------------------------------------------------------
// Online-softmax update for one 16x32 score tile held as two C/D accumulators.
// Writes exp-weights (bf16) into the wave-private LDS staging tile.
// ---------------------------------------------------------------------------
__device__ __forceinline__ void softmax_update(
    const v8f& c0, const v8f& c1,
    v8f& o0, v8f& o1, v8f& o2, v8f& o3,
    float mrun[8], float lrun[8],
    const float* __restrict__ R, const int* __restrict__ mask,
    int tbase, int rowoff, int sA, int sB, int nloc,
    bf16_t* __restrict__ pdst) {
  const float scale = 0.125f;                      // 1/sqrt(64)
#pragma unroll
  for (int r = 0; r < 8; ++r) {
    const int t = tbase + r;
    const size_t rowT = (size_t)t * T_;
    float x0 = c0[r] * scale + R[rowT + sA];
    float x1 = c1[r] * scale + R[rowT + sB];
    if (mask[rowT + sA] == 1) x0 = NEG_BIG;
    if (mask[rowT + sB] == 1) x1 = NEG_BIG;

    float mx = fmaxf(x0, x1);
    mx = fmaxf(mx, __shfl_xor(mx, 1, 16));
    mx = fmaxf(mx, __shfl_xor(mx, 2, 16));
    mx = fmaxf(mx, __shfl_xor(mx, 4, 16));
    mx = fmaxf(mx, __shfl_xor(mx, 8, 16));

    const float mnew  = fmaxf(mrun[r], mx);
    const float alpha = __expf(mrun[r] - mnew);
    const float p0 = __expf(x0 - mnew);
    const float p1 = __expf(x1 - mnew);

    float rs = p0 + p1;
    rs += __shfl_xor(rs, 1, 16);
    rs += __shfl_xor(rs, 2, 16);
    rs += __shfl_xor(rs, 4, 16);
    rs += __shfl_xor(rs, 8, 16);

    lrun[r] = lrun[r] * alpha + rs;
    mrun[r] = mnew;
    o0[r] *= alpha; o1[r] *= alpha; o2[r] *= alpha; o3[r] *= alpha;

    const int m = r + rowoff;
    pdst[m * 32 + nloc]      = (bf16_t)p0;
    pdst[m * 32 + 16 + nloc] = (bf16_t)p1;
  }
}

// ---------------------------------------------------------------------------
// Flash-style attention. blockDim = 128 (4 independent waves).
// grid: (T/128, B*H). Each wave owns a 32-row Q tile, streams 32-wide K/V
// tiles; V B-frags are shared by two P@V WMMAs (halves K/V L2 re-reads).
// Writes y (pre output-projection) as bf16 [B*T, C].
// ---------------------------------------------------------------------------
__global__ __launch_bounds__(128)
void attention_kernel(const bf16_t* __restrict__ qb, const bf16_t* __restrict__ kb,
                      const bf16_t* __restrict__ vtb,
                      const float* __restrict__ rel, const int* __restrict__ mask,
                      bf16_t* __restrict__ yb) {
  const int lane = threadIdx.x & 31;
  const int wave = threadIdx.x >> 5;
  const int bh = blockIdx.y;            // 0..31
  const int b  = bh >> 4;
  const int h  = bh & 15;
  const int t0 = (blockIdx.x * 4 + wave) * 32;

  const bf16_t* Q  = qb  + (size_t)bh * T_ * D_;
  const bf16_t* K  = kb  + (size_t)bh * T_ * D_;
  const bf16_t* Vt = vtb + (size_t)bh * D_ * T_;
  const float*  R  = rel + (size_t)h * T_ * T_;

  __shared__ __align__(32) bf16_t plds[4][2][16 * 32];

  // Q tile: 2 row-halves x 2 d-halves, loaded once.
  const v16bf aq00 = load_a_frag(Q, D_, t0,      0,  lane);
  const v16bf aq01 = load_a_frag(Q, D_, t0,      32, lane);
  const v16bf aq10 = load_a_frag(Q, D_, t0 + 16, 0,  lane);
  const v16bf aq11 = load_a_frag(Q, D_, t0 + 16, 32, lane);

  v8f o[2][4];
  float mrun[2][8], lrun[2][8];
#pragma unroll
  for (int ti = 0; ti < 2; ++ti) {
#pragma unroll
    for (int j = 0; j < 4; ++j) o[ti][j] = v8f_zero();
#pragma unroll
    for (int r = 0; r < 8; ++r) { mrun[ti][r] = NEG_BIG; lrun[ti][r] = 0.f; }
  }

  const int nloc   = lane & 15;
  const int rowoff = 8 * (lane >> 4);

  for (int s0 = 0; s0 < T_; s0 += 32) {
    // prefetch next rel_pos tile (one cacheline probe per lane)
    if (s0 + 32 < T_)
      __builtin_prefetch(&R[(size_t)(t0 + rowoff) * T_ + s0 + 32 + nloc], 0, 0);

    // ---- scores: two 16x32 tiles, d-reduction via 2 WMMAs each half ----
    v16bf k0a = load_b_frag(K, D_, s0,      0,  lane);
    v16bf k0b = load_b_frag(K, D_, s0,      32, lane);
    v16bf k1a = load_b_frag(K, D_, s0 + 16, 0,  lane);
    v16bf k1b = load_b_frag(K, D_, s0 + 16, 32, lane);

    v8f c00 = v8f_zero();                  // rows t0..t0+15, cols s0..s0+15
    c00 = wmma_bf16(aq00, k0a, c00);
    c00 = wmma_bf16(aq01, k0b, c00);
    v8f c01 = v8f_zero();                  // rows t0..t0+15, cols s0+16..s0+31
    c01 = wmma_bf16(aq00, k1a, c01);
    c01 = wmma_bf16(aq01, k1b, c01);
    v8f c10 = v8f_zero();                  // rows t0+16..t0+31
    c10 = wmma_bf16(aq10, k0a, c10);
    c10 = wmma_bf16(aq11, k0b, c10);
    v8f c11 = v8f_zero();
    c11 = wmma_bf16(aq10, k1a, c11);
    c11 = wmma_bf16(aq11, k1b, c11);

    const int sA = s0 + nloc;
    const int sB = s0 + 16 + nloc;

    softmax_update(c00, c01, o[0][0], o[0][1], o[0][2], o[0][3],
                   mrun[0], lrun[0], R, mask,
                   t0 + rowoff, rowoff, sA, sB, nloc, plds[wave][0]);
    softmax_update(c10, c11, o[1][0], o[1][1], o[1][2], o[1][3],
                   mrun[1], lrun[1], R, mask,
                   t0 + 16 + rowoff, rowoff, sA, sB, nloc, plds[wave][1]);

    // same-wave DS ordering: stores visible before the A-frag reads
    asm volatile("s_wait_dscnt 0" ::: "memory");

    const v16bf pa0 = load_a_frag(plds[wave][0], 32, 0, 0, lane);
    const v16bf pa1 = load_a_frag(plds[wave][1], 32, 0, 0, lane);

    // ---- P @ V : each V B-frag feeds both row tiles ----
#pragma unroll
    for (int j = 0; j < 4; ++j) {
      v16bf vb = load_b_frag(Vt, T_, j * 16, s0, lane);
      o[0][j] = wmma_bf16(pa0, vb, o[0][j]);
      o[1][j] = wmma_bf16(pa1, vb, o[1][j]);
    }

    asm volatile("s_wait_dscnt 0" ::: "memory");  // WAR vs next tile's stores
  }

  // ---- normalize and write y as [B*T, C] bf16 ----
#pragma unroll
  for (int ti = 0; ti < 2; ++ti) {
#pragma unroll
    for (int r = 0; r < 8; ++r) {
      const float inv = 1.0f / lrun[ti][r];
      const int t = t0 + ti * 16 + r + rowoff;
      const size_t base = ((size_t)b * T_ + t) * C_ + h * D_ + nloc;
      yb[base +  0] = (bf16_t)(o[ti][0][r] * inv);
      yb[base + 16] = (bf16_t)(o[ti][1][r] * inv);
      yb[base + 32] = (bf16_t)(o[ti][2][r] * inv);
      yb[base + 48] = (bf16_t)(o[ti][3][r] * inv);
    }
  }
}

// ---------------------------------------------------------------------------
// Output projection: out = y @ Wp^T + bp, fp32 result. Same 32x64 tiling.
// ---------------------------------------------------------------------------
__global__ __launch_bounds__(128)
void proj_gemm_kernel(const bf16_t* __restrict__ yb, const bf16_t* __restrict__ wp,
                      const float* __restrict__ bp, float* __restrict__ out) {
  const int lane = threadIdx.x & 31;
  const int wave = threadIdx.x >> 5;
  const int n0 = blockIdx.x * 64;
  const int m0 = blockIdx.y * 128 + wave * 32;

  v8f acc[2][4];
#pragma unroll
  for (int mi = 0; mi < 2; ++mi)
#pragma unroll
    for (int nj = 0; nj < 4; ++nj) acc[mi][nj] = v8f_zero();

  for (int k0 = 0; k0 < C_; k0 += 32) {
    v16bf a0 = load_a_frag(yb, C_, m0,      k0, lane);
    v16bf a1 = load_a_frag(yb, C_, m0 + 16, k0, lane);
#pragma unroll
    for (int nj = 0; nj < 4; ++nj) {
      v16bf b = load_b_frag(wp, C_, n0 + nj * 16, k0, lane);
      acc[0][nj] = wmma_bf16(a0, b, acc[0][nj]);
      acc[1][nj] = wmma_bf16(a1, b, acc[1][nj]);
    }
  }

#pragma unroll
  for (int nj = 0; nj < 4; ++nj) {
    const int n = n0 + nj * 16 + (lane & 15);
    const float bval = bp[n];
#pragma unroll
    for (int mi = 0; mi < 2; ++mi) {
#pragma unroll
      for (int r = 0; r < 8; ++r) {
        const int m = m0 + mi * 16 + r + 8 * (lane >> 4);
        out[(size_t)m * C_ + n] = acc[mi][nj][r] + bval;
      }
    }
  }
}

// ---------------------------------------------------------------------------
// Host launcher
// ---------------------------------------------------------------------------
extern "C" void kernel_launch(void* const* d_in, const int* in_sizes, int n_in,
                              void* d_out, int out_size, void* d_ws, size_t ws_size,
                              hipStream_t stream) {
  (void)in_sizes; (void)n_in; (void)out_size; (void)ws_size;

  const float* x    = (const float*)d_in[0];
  const int*   mask = (const int*)d_in[1];
  const float* rel  = (const float*)d_in[2];
  const float* Wq   = (const float*)d_in[3];
  const float* bq   = (const float*)d_in[4];
  const float* Wk   = (const float*)d_in[5];
  const float* bk   = (const float*)d_in[6];
  const float* Wv   = (const float*)d_in[7];
  const float* bv   = (const float*)d_in[8];
  const float* Wp   = (const float*)d_in[9];
  const float* bp   = (const float*)d_in[10];
  float* out = (float*)d_out;

  // workspace carve-out (all 256B aligned)
  size_t off = 0;
  auto alloc = [&](size_t bytes) -> void* {
    void* p = (char*)d_ws + off;
    off += (bytes + 255) & ~(size_t)255;
    return p;
  };
  bf16_t* xb  = (bf16_t*)alloc((size_t)M_ * C_ * 2);       // x in bf16
  bf16_t* wqb = (bf16_t*)alloc((size_t)C_ * C_ * 2);
  bf16_t* wkb = (bf16_t*)alloc((size_t)C_ * C_ * 2);
  bf16_t* wvb = (bf16_t*)alloc((size_t)C_ * C_ * 2);
  bf16_t* wpb = (bf16_t*)alloc((size_t)C_ * C_ * 2);
  bf16_t* qb  = (bf16_t*)alloc((size_t)M_ * C_ * 2);       // [b,h,t,d]
  bf16_t* kbx = (bf16_t*)alloc((size_t)M_ * C_ * 2);       // [b,h,t,d]
  bf16_t* vtb = (bf16_t*)alloc((size_t)M_ * C_ * 2);       // [b,h,d,t]
  bf16_t* yb  = (bf16_t*)alloc((size_t)M_ * C_ * 2);       // [B*T, C]

  // 1) precision conversion fp32 -> bf16
  cvt_f32_bf16<<<4096, 256, 0, stream>>>(x,  xb,  M_ * C_);
  cvt_f32_bf16<<<2048, 256, 0, stream>>>(Wq, wqb, C_ * C_);
  cvt_f32_bf16<<<2048, 256, 0, stream>>>(Wk, wkb, C_ * C_);
  cvt_f32_bf16<<<2048, 256, 0, stream>>>(Wv, wvb, C_ * C_);
  cvt_f32_bf16<<<2048, 256, 0, stream>>>(Wp, wpb, C_ * C_);

  // 2) QKV projections (WMMA bf16, 32x64 tiles)
  {
    dim3 grid(C_ / 64, M_ / 128, 3);
    qkv_gemm_kernel<<<grid, 128, 0, stream>>>(xb, wqb, wkb, wvb, bq, bk, bv,
                                              qb, kbx, vtb);
  }

  // 3) flash attention with rel_pos + mask (WMMA bf16, online softmax)
  {
    dim3 grid(T_ / 128, B_ * H_);
    attention_kernel<<<grid, 128, 0, stream>>>(qb, kbx, vtb, rel, mask, yb);
  }

  // 4) output projection (WMMA bf16, fp32 out, 32x64 tiles)
  {
    dim3 grid(C_ / 64, M_ / 128);
    proj_gemm_kernel<<<grid, 128, 0, stream>>>(yb, wpb, bp, out);
  }
}
